// GatedGCNConv_31404800868644
// MI455X (gfx1250) — matrix operand order repaired and checked by reading, not compiled, not run on hardware
//
#include <hip/hip_runtime.h>
#include <hip/hip_bf16.h>

#define NN 100000
#define NE 600000
#define DIM 128
#define EDIM 16
#define MTILES (NN / 16)   // 6250 exactly

typedef __attribute__((ext_vector_type(16))) __bf16 v16bf;
typedef __attribute__((ext_vector_type(8)))  float  v8f;

__device__ __forceinline__ float sigmoidf(float v) {
    return 1.0f / (1.0f + __expf(-v));
}

// ---------------------------------------------------------------------------
// Kernel 1: pack the four 128x128 fp32 weights into bf16 B-matrix fragments.
// Fragment layout (contiguous K order, same as A fragments): lane&15 = N column,
// lane>>4 selects K half (+16), dword j holds K = base + 2j, 2j+1.
// Linear index = ((((w*8 + nt)*4 + kt)*32) + lane)*8 + j  -> one dword each.
// ---------------------------------------------------------------------------
__global__ __launch_bounds__(256) void pack_weights_kernel(
    const float* __restrict__ Aw, const float* __restrict__ Bw,
    const float* __restrict__ Cw, const float* __restrict__ Dw,
    unsigned* __restrict__ packed)
{
    int idx  = blockIdx.x * 256 + threadIdx.x;      // 32768 total
    int j    = idx & 7;
    int lane = (idx >> 3) & 31;
    int kt   = (idx >> 8) & 3;
    int nt   = (idx >> 10) & 7;
    int w    = (idx >> 13) & 3;
    const float* W = (w == 0) ? Aw : (w == 1) ? Bw : (w == 2) ? Cw : Dw;
    int ncol = nt * 16 + (lane & 15);
    int k0   = kt * 32 + ((lane >> 4) * 16) + 2 * j;
    // y = x @ W.T  =>  B[k][n] = W[n][k]
    __bf16 b0 = (__bf16)W[ncol * DIM + k0];
    __bf16 b1 = (__bf16)W[ncol * DIM + k0 + 1];
    unsigned u0 = (unsigned)__builtin_bit_cast(unsigned short, b0);
    unsigned u1 = (unsigned)__builtin_bit_cast(unsigned short, b1);
    packed[idx] = u0 | (u1 << 16);
}

// ---------------------------------------------------------------------------
// Kernel 2: zero agg + BN accumulators (must run every launch).
// ---------------------------------------------------------------------------
__global__ __launch_bounds__(256) void zero_kernel(
    float* __restrict__ agg, float* __restrict__ sums, float* __restrict__ sumsq)
{
    int idx = blockIdx.x * 256 + threadIdx.x;
    if (idx < (NN * DIM) / 4) {
        ((float4*)agg)[idx] = make_float4(0.f, 0.f, 0.f, 0.f);
    }
    if (idx < DIM) { sums[idx] = 0.f; sumsq[idx] = 0.f; }
}

// ---------------------------------------------------------------------------
// Kernel 3: fused node GEMM via bf16 WMMA. One wave = one 16-node tile.
// Computes Ax, Bx, Cx and sigmoid(Dx). 128 v_wmma per wave.
// B fragments are software-pipelined (bcur/bnext) so each fragment load is
// in flight while the previous WMMA executes — avoids load->wait0->wmma
// serialization seen in the unpipelined schedule.
// ---------------------------------------------------------------------------
__global__ __launch_bounds__(256) void node_gemm_kernel(
    const float* __restrict__ x, const unsigned* __restrict__ packedW,
    const float* __restrict__ Ab, const float* __restrict__ Bb,
    const float* __restrict__ Cb, const float* __restrict__ Db,
    float* __restrict__ Ax, float* __restrict__ Bx,
    float* __restrict__ Cx, float* __restrict__ SD)
{
    const int lane  = threadIdx.x & 31;
    const int wave  = threadIdx.x >> 5;
    const int mtile = blockIdx.x * 8 + wave;
    if (mtile >= MTILES) return;                    // wave-uniform: EXEC stays full
    const int half = lane >> 4;
    const int m    = mtile * 16 + (lane & 15);
    const float2* xrow = (const float2*)(x + (size_t)m * DIM);

    // A fragments: 16x32 bf16. lane&15 = M row; lanes 16-31 take K+16;
    // dword j holds K = base + 2j, 2j+1 (matches packed B).
    v16bf afrag[4];
#pragma unroll
    for (int kt = 0; kt < 4; ++kt) {
#pragma unroll
        for (int j = 0; j < 8; ++j) {
            int k0 = kt * 32 + half * 16 + 2 * j;
            float2 p = xrow[k0 >> 1];
            afrag[kt][2 * j]     = (__bf16)p.x;
            afrag[kt][2 * j + 1] = (__bf16)p.y;
        }
    }

    const float* biases[4] = {Ab, Bb, Cb, Db};
    float*       outs[4]   = {Ax, Bx, Cx, SD};

    // Per-lane fragment pointer: fragment f lives at v16bf index f*32 + lane.
    const v16bf* Bp = ((const v16bf*)packedW) + lane;
    v16bf bcur = Bp[0];

#pragma unroll
    for (int tile = 0; tile < 32; ++tile) {         // tile = w*8 + nt
        const int w  = tile >> 3;
        const int nt = tile & 7;
        const int ncol = nt * 16 + (lane & 15);
        const float bv = biases[w][ncol];
        v8f acc = {bv, bv, bv, bv, bv, bv, bv, bv};
#pragma unroll
        for (int kt = 0; kt < 4; ++kt) {
            const int nf = tile * 4 + kt + 1;       // next fragment index
            v16bf bnext = (nf < 128) ? Bp[(size_t)nf * 32] : bcur;
            acc = __builtin_amdgcn_wmma_f32_16x16x32_bf16(
                false, afrag[kt], false, bcur, (short)0, acc, false, false);
            bcur = bnext;
        }
        // C/D layout: VGPR r, lanes 0-15 -> M=r, lanes 16-31 -> M=r+8
        float* outp = outs[w];
        if (w < 3) {
#pragma unroll
            for (int r = 0; r < 8; ++r) {
                int row = mtile * 16 + half * 8 + r;
                outp[(size_t)row * DIM + ncol] = acc[r];
            }
        } else {
#pragma unroll
            for (int r = 0; r < 8; ++r) {
                int row = mtile * 16 + half * 8 + r;
                outp[(size_t)row * DIM + ncol] = sigmoidf(acc[r]);
            }
        }
    }
}

// ---------------------------------------------------------------------------
// Kernel 4: edge phase. 8 edges/block, 32 lanes/edge, 4 cols/lane.
// Gathers hit L2 (Ax/Bx/Cx = 153.6 MB < 192 MB L2); scatter = fp32 atomics.
// ---------------------------------------------------------------------------
__global__ __launch_bounds__(256) void edge_kernel(
    const long long* __restrict__ ei, const float* __restrict__ ea,
    const float* __restrict__ Ew, const float* __restrict__ Eb,
    const float* __restrict__ Ax, const float* __restrict__ Bx,
    const float* __restrict__ Cx, float* __restrict__ agg)
{
    __shared__ float sEwT[EDIM * DIM];   // transposed: [k][col], float4-friendly
    __shared__ float sEb[DIM];
    __shared__ float sEa[8 * EDIM];

    const int t = threadIdx.x;
#pragma unroll
    for (int i = 0; i < 8; ++i) {
        int idx = i * 256 + t;           // 2048 entries
        int k = idx >> 7, col = idx & 127;
        sEwT[k * DIM + col] = Ew[col * EDIM + k];
    }
    if (t < DIM) sEb[t] = Eb[t];
    const int ebase = blockIdx.x * 8;
    if (t < 128) {
        int e = ebase + (t >> 4);
        sEa[t] = (e < NE) ? ea[(size_t)e * EDIM + (t & 15)] : 0.0f;
    }
    __syncthreads();

    const int slot = t >> 5, lane = t & 31;
    const int e = ebase + slot;
    if (e >= NE) return;
    const long long src = ei[e];
    const long long dst = ei[NE + e];
    const int c0 = lane * 4;

    float4 ex = make_float4(sEb[c0], sEb[c0 + 1], sEb[c0 + 2], sEb[c0 + 3]);
    const float4* sW4 = (const float4*)sEwT;
#pragma unroll
    for (int k = 0; k < EDIM; ++k) {
        float  a  = sEa[slot * EDIM + k];
        float4 wv = sW4[k * 32 + lane];
        ex.x += a * wv.x; ex.y += a * wv.y; ex.z += a * wv.z; ex.w += a * wv.w;
    }

    const float4* Ax4 = (const float4*)Ax;
    const float4* Bx4 = (const float4*)Bx;
    const float4* Cx4 = (const float4*)Cx;
    size_t so = (size_t)src * 32 + lane;
    size_t dofs = (size_t)dst * 32 + lane;
    float4 av = Ax4[so], bv = Bx4[so], cv = Cx4[dofs];

    float4 g;
    g.x = sigmoidf(bv.x + cv.x + ex.x);
    g.y = sigmoidf(bv.y + cv.y + ex.y);
    g.z = sigmoidf(bv.z + cv.z + ex.z);
    g.w = sigmoidf(bv.w + cv.w + ex.w);

    float* ad = agg + (size_t)dst * DIM + c0;
    atomicAdd(ad + 0, av.x * g.x);
    atomicAdd(ad + 1, av.y * g.y);
    atomicAdd(ad + 2, av.z * g.z);
    atomicAdd(ad + 3, av.w * g.w);
}

// ---------------------------------------------------------------------------
// Kernel 5: gated residual (in place into agg) + BN column sum / sumsq.
// 256 rows per block; thread col = t&127; LDS pair-reduce then global atomics.
// ---------------------------------------------------------------------------
__global__ __launch_bounds__(256) void stats_kernel(
    const float* __restrict__ x, const float* __restrict__ SD,
    float* __restrict__ agg, float* __restrict__ sums, float* __restrict__ sumsq)
{
    __shared__ float s1[256], s2[256];
    const int t = threadIdx.x;
    const int col = t & 127;
    const int base = blockIdx.x * 256;
    float sm = 0.f, sq = 0.f;
    for (int r = base + (t >> 7); r < base + 256; r += 2) {
        if (r < NN) {
            size_t i = (size_t)r * DIM + col;
            float pre = agg[i] * SD[i] + x[i];
            agg[i] = pre;
            sm += pre; sq += pre * pre;
        }
    }
    s1[t] = sm; s2[t] = sq;
    __syncthreads();
    if (t < 128) {
        atomicAdd(&sums[col],  s1[t] + s1[t + 128]);
        atomicAdd(&sumsq[col], s2[t] + s2[t + 128]);
    }
}

// ---------------------------------------------------------------------------
// Kernel 6: BatchNorm (biased var) + ReLU -> d_out.
// ---------------------------------------------------------------------------
__global__ __launch_bounds__(256) void final_kernel(
    const float* __restrict__ pre, const float* __restrict__ sums,
    const float* __restrict__ sumsq, const float* __restrict__ gamma,
    const float* __restrict__ beta, float* __restrict__ out)
{
    int idx = blockIdx.x * 256 + threadIdx.x;       // float4 units
    if (idx >= (NN * DIM) / 4) return;
    const int cq = (idx & 31) * 4;                  // 32 float4 per row
    const float invN = 1.0f / (float)NN;
    float4 p = ((const float4*)pre)[idx];
    float pv[4] = {p.x, p.y, p.z, p.w};
    float ov[4];
#pragma unroll
    for (int c = 0; c < 4; ++c) {
        int col = cq + c;
        float mean = sums[col] * invN;
        float var  = sumsq[col] * invN - mean * mean;
        float inv  = rsqrtf(var + 1e-5f);
        float v    = gamma[col] * (pv[c] - mean) * inv + beta[col];
        ov[c] = fmaxf(v, 0.0f);
    }
    ((float4*)out)[idx] = make_float4(ov[0], ov[1], ov[2], ov[3]);
}

// ---------------------------------------------------------------------------
extern "C" void kernel_launch(void* const* d_in, const int* in_sizes, int n_in,
                              void* d_out, int out_size, void* d_ws, size_t ws_size,
                              hipStream_t stream) {
    const float*     x    = (const float*)d_in[0];
    const long long* ei   = (const long long*)d_in[1];   // int64 [2, E]
    const float*     ea   = (const float*)d_in[2];
    const float*     Aw   = (const float*)d_in[3];
    const float*     Ab   = (const float*)d_in[4];
    const float*     Bw   = (const float*)d_in[5];
    const float*     Bb   = (const float*)d_in[6];
    const float*     Cw   = (const float*)d_in[7];
    const float*     Cb   = (const float*)d_in[8];
    const float*     Dw   = (const float*)d_in[9];
    const float*     Db   = (const float*)d_in[10];
    const float*     Ew   = (const float*)d_in[11];
    const float*     Eb   = (const float*)d_in[12];
    const float*     gam  = (const float*)d_in[13];
    const float*     bet  = (const float*)d_in[14];
    float*           out  = (float*)d_out;

    const size_t NF = (size_t)NN * DIM;              // 12.8M floats
    char* ws = (char*)d_ws;
    unsigned* packedW = (unsigned*)ws;               // 4*128*128*2 B = 131072
    float* Ax  = (float*)(ws + 131072);
    float* Bx  = Ax  + NF;
    float* Cx  = Bx  + NF;
    float* SD  = Cx  + NF;
    float* agg = SD  + NF;
    float* sums  = agg + NF;
    float* sumsq = sums + DIM;

    pack_weights_kernel<<<128, 256, 0, stream>>>(Aw, Bw, Cw, Dw, packedW);
    zero_kernel<<<(int)((NF / 4 + 255) / 256), 256, 0, stream>>>(agg, sums, sumsq);
    node_gemm_kernel<<<(MTILES + 7) / 8, 256, 0, stream>>>(
        x, packedW, Ab, Bb, Cb, Db, Ax, Bx, Cx, SD);
    edge_kernel<<<(NE + 7) / 8, 256, 0, stream>>>(ei, ea, Ew, Eb, Ax, Bx, Cx, agg);
    stats_kernel<<<(NN + 255) / 256, 256, 0, stream>>>(x, SD, agg, sums, sumsq);
    final_kernel<<<(int)((NF / 4 + 255) / 256), 256, 0, stream>>>(
        agg, sums, sumsq, gam, bet, out);
}
